// ModulatedConv2d_76089640616441
// MI455X (gfx1250) — matrix operand order repaired and checked by reading, not compiled, not run on hardware
//
#include <hip/hip_runtime.h>
#include <hip/hip_bf16.h>

typedef __bf16 bf16_t;
typedef __attribute__((ext_vector_type(16))) __bf16 v16bf;
typedef __attribute__((ext_vector_type(8)))  __bf16 v8bf;
typedef __attribute__((ext_vector_type(8)))  float  v8f;

#define BB  16
#define IC  512
#define OC  512
#define SD  512
#define HH  64
#define WW  64
#define HWN 4096

#define A_STRIDE 40   // 32 bf16 + 8 pad (80B rows -> conflict-free b128 lane reads)
#define B_STRIDE 40

// ---------------- kernel 1: style modulation s[b][ic] ----------------
__global__ __launch_bounds__(512) void k_style(
    const float* __restrict__ style,
    const float* __restrict__ mod_weight,
    const float* __restrict__ mod_bias,
    float* __restrict__ sbuf)
{
    const int b  = blockIdx.x;
    const int ic = threadIdx.x;
    const float mod_scale = 0.04419417382f;   // 1/sqrt(512)
    const float* st = style + b * SD;
    const float* mw = mod_weight + (size_t)ic * SD;
    float acc = 0.f;
    for (int d = 0; d < SD; ++d) acc += st[d] * mw[d];
    sbuf[b * IC + ic] = acc * mod_scale + mod_bias[ic];
}

// ---- kernel 2: weight transform [oc][ic][t] -> [t][oc][ic] + squared sums ----
__global__ __launch_bounds__(256) void k_wt(
    const float* __restrict__ weight,
    float* __restrict__ wT,          // [9][OC*IC]
    float* __restrict__ wsq)         // [OC*IC]
{
    const int idx = blockIdx.x * 256 + threadIdx.x;   // oc*IC + ic
    const float* wp = weight + (size_t)idx * 9;
    float a = 0.f;
    #pragma unroll
    for (int t = 0; t < 9; ++t) {
        const float v = wp[t];
        a += v * v;
        wT[(size_t)t * (OC * IC) + idx] = v;
    }
    wsq[idx] = a;
}

// ---------------- kernel 3: demodulation factor ----------------
__global__ __launch_bounds__(256) void k_demod(
    const float* __restrict__ wsq, const float* __restrict__ sbuf,
    float* __restrict__ dbuf)
{
    const int b  = blockIdx.y;
    const int oc = blockIdx.x * 256 + threadIdx.x;
    const float* wq = wsq + (size_t)oc * IC;
    const float* sp = sbuf + b * IC;
    float a = 0.f;
    for (int ic = 0; ic < IC; ++ic) { float sv = sp[ic]; a += wq[ic] * sv * sv; }
    dbuf[b * OC + oc] = rsqrtf(a * (1.0f / 4608.0f) + 1e-8f);  // conv_scale^2 = 1/4608
}

// -------- kernel 4: implicit-GEMM modulated conv, 128x128 tile, bf16 WMMA --------
__global__ __launch_bounds__(256) void k_modconv(
    const float* __restrict__ input,    // [B,IC,64,64]
    const float* __restrict__ wT,       // [9][OC][IC]
    const float* __restrict__ sbuf,     // [B,IC]
    const float* __restrict__ dbuf,     // [B,OC]
    float* __restrict__ out)            // [B,OC,64,64]
{
    __shared__ __align__(16) bf16_t Asm[2][128 * A_STRIDE];   // [m][k] modulated weights
    __shared__ __align__(16) bf16_t Bsm[2][128 * B_STRIDE];   // [n][k] im2col input

    const int b   = blockIdx.z;
    const int oc0 = blockIdx.y << 7;    // 128-wide OC tile
    const int n0  = blockIdx.x << 7;    // 128-wide HW tile

    const int tid  = threadIdx.x;
    const int lane = tid & 31;
    const int wave = tid >> 5;          // 0..7
    const int wm   = wave >> 1;         // m-pair 0..3 (32 rows each)
    const int wn   = wave & 1;          // n-half 0..1 (64 cols each)
    const int half = lane >> 4;
    const int l16  = lane & 15;

    // staging assignments
    const int am  = tid >> 1;           // 0..127 (A row = oc)
    const int ak0 = (tid & 1) << 4;     // 0/16   (16 contiguous ic)
    const int bkk = tid >> 3;           // 0..31  (B k = ic)
    const int bn  = (tid & 7) << 4;     // 0..112 (16 contiguous n)

    const float conv_scale = 0.01473139127f;   // 1/sqrt(512*9)
    const float dscale = dbuf[b * OC + oc0 + am] * conv_scale;
    const float* __restrict__ srow = sbuf + b * IC;

    v8f acc[2][4] = {};

    // stage k-step s (t = s/16, ic-chunk = (s%16)*32) into buffers As/Bs
    auto stage = [&](int s, bf16_t* As, bf16_t* Bs) {
        const int t   = s >> 4;
        const int icc = (s & 15) << 5;
        // ---- A: modulate + convert weights (contiguous wT reads) ----
        {
            const float* wp = wT + (size_t)t * (OC * IC) + (size_t)(oc0 + am) * IC + icc + ak0;
            const float* sp = srow + icc + ak0;
            bf16_t* ap = As + am * A_STRIDE + ak0;
            #pragma unroll
            for (int j = 0; j < 16; ++j)
                ap[j] = (bf16_t)(wp[j] * sp[j] * dscale);
        }
        // ---- B: shifted/masked input (im2col) ----
        {
            const int dh = t / 3 - 1;
            const int dw = t % 3 - 1;
            const int h  = ((n0 + bn) >> 6) + dh;       // constant over 16-wide n run
            const int w0 = ((n0 + bn) & 63) + dw;
            const bool hok = (h >= 0) && (h < HH);
            const float* ip = input + ((size_t)(b * IC + icc + bkk) * HH + h) * WW;
            bf16_t* bp = Bs + bn * B_STRIDE + bkk;
            #pragma unroll
            for (int j = 0; j < 16; ++j) {
                const int w = w0 + j;
                const float v = (hok && w >= 0 && w < WW) ? ip[w] : 0.0f;
                bp[j * B_STRIDE] = (bf16_t)v;
            }
        }
    };

    // 8 WMMAs on buffers As/Bs
    auto compute = [&](const bf16_t* As, const bf16_t* Bs) {
        v16bf afrag[2];
        #pragma unroll
        for (int g = 0; g < 2; ++g) {
            // A frag: lane(half,l16): k = half*8+{0..7}, 16+half*8+{0..7}
            const v8bf* ap = (const v8bf*)(As + ((wm << 5) + (g << 4) + l16) * A_STRIDE + (half << 3));
            const v8bf lo = ap[0];
            const v8bf hi = ap[2];   // +16 bf16
            #pragma unroll
            for (int i = 0; i < 8; ++i) { afrag[g][i] = lo[i]; afrag[g][i + 8] = hi[i]; }
        }
        #pragma unroll
        for (int f = 0; f < 4; ++f) {
            // B frag: lane(half, n=l16): k = half*8+{0..7}, 16+half*8+{0..7}
            const v8bf* bp = (const v8bf*)(Bs + ((wn << 6) + (f << 4) + l16) * B_STRIDE + (half << 3));
            const v8bf lo = bp[0];
            const v8bf hi = bp[2];
            v16bf bfrag;
            #pragma unroll
            for (int i = 0; i < 8; ++i) { bfrag[i] = lo[i]; bfrag[i + 8] = hi[i]; }
            #pragma unroll
            for (int g = 0; g < 2; ++g)
                acc[g][f] = __builtin_amdgcn_wmma_f32_16x16x32_bf16(
                    false, afrag[g], false, bfrag, (short)0, acc[g][f], false, false);
        }
    };

    const int NSTEP = 9 * 16;   // 144 k-steps of 32
    stage(0, Asm[0], Bsm[0]);
    for (int s = 0; s < NSTEP; ++s) {
        __syncthreads();                       // cur buffer staged; prev reads done
        const int cur = s & 1;
        if (s + 1 < NSTEP)
            stage(s + 1, Asm[cur ^ 1], Bsm[cur ^ 1]);   // overlaps with WMMA below
        compute(Asm[cur], Bsm[cur]);
    }

    // ---- epilogue: C/D layout lane n = l16, VGPR r -> m = wm*32 + g*16 + half*8 + r ----
    #pragma unroll
    for (int g = 0; g < 2; ++g) {
        #pragma unroll
        for (int f = 0; f < 4; ++f) {
            const int n = n0 + (wn << 6) + (f << 4) + l16;
            float* op = out + (size_t)(b * OC + oc0 + (wm << 5) + (g << 4) + (half << 3)) * HWN + n;
            #pragma unroll
            for (int r = 0; r < 8; ++r)
                op[(size_t)r * HWN] = acc[g][f][r];
        }
    }
}

extern "C" void kernel_launch(void* const* d_in, const int* in_sizes, int n_in,
                              void* d_out, int out_size, void* d_ws, size_t ws_size,
                              hipStream_t stream) {
    (void)in_sizes; (void)n_in; (void)out_size; (void)ws_size;
    const float* input      = (const float*)d_in[0];
    const float* style      = (const float*)d_in[1];
    const float* weight     = (const float*)d_in[2];
    const float* mod_weight = (const float*)d_in[3];
    const float* mod_bias   = (const float*)d_in[4];
    float* out = (float*)d_out;

    float* ws   = (float*)d_ws;
    float* sbuf = ws;                          // [B,IC]      8192 floats
    float* dbuf = sbuf + BB * IC;              // [B,OC]      8192 floats
    float* wsq  = dbuf + BB * OC;              // [OC*IC]     262144 floats
    float* wT   = wsq + OC * IC;               // [9][OC*IC]  2359296 floats (~10.6 MB total)

    k_style<<<dim3(BB),              dim3(SD),  0, stream>>>(style, mod_weight, mod_bias, sbuf);
    k_wt   <<<dim3((OC * IC) / 256), dim3(256), 0, stream>>>(weight, wT, wsq);
    k_demod<<<dim3(OC / 256, BB),    dim3(256), 0, stream>>>(wsq, sbuf, dbuf);
    k_modconv<<<dim3(HWN / 128, OC / 128, BB), dim3(256), 0, stream>>>(input, wT, sbuf, dbuf, out);
}
